// Attention_42004780155294
// MI455X (gfx1250) — matrix operand order repaired and checked by reading, not compiled, not run on hardware
//
#include <hip/hip_runtime.h>
#include <hip/hip_bf16.h>

// ---------------------------------------------------------------------------
// Types for CDNA5 WMMA (wave32): v_wmma_f32_16x16x32_bf16
// ---------------------------------------------------------------------------
typedef __attribute__((ext_vector_type(16))) __bf16 v16bf;
typedef __attribute__((ext_vector_type(8)))  __bf16 v8bf;
typedef __attribute__((ext_vector_type(4)))  __bf16 v4bf;
typedef __attribute__((ext_vector_type(8)))  float  v8f;

// Problem constants (from reference)
#define BSZ   2
#define TLEN  2048
#define CDIM  1024
#define NH    16
#define HD    64
#define RANK  32
#define MROWS (BSZ * TLEN)     // 4096
#define C3    (3 * CDIM)       // 3072
#define LORA_SCALE 0.5f

// ---------------------------------------------------------------------------
// CDNA5 async copy global->LDS (no VGPR round trip, tracked by ASYNCcnt).
// LDS address operand: flat generic pointers to __shared__ map to LDS via
// addr[31:0] (ISA 10.2), so truncation yields the VDST byte address.
// ---------------------------------------------------------------------------
__device__ __forceinline__ unsigned lds_off32(const void* p) {
  return (unsigned)(unsigned long long)p;
}
__device__ __forceinline__ void async_copy_b128(const void* gsrc, void* ldst) {
  asm volatile("global_load_async_to_lds_b128 %0, %1, off"
               :: "v"(lds_off32(ldst)), "v"(gsrc) : "memory");
}
__device__ __forceinline__ void wait_async0() {
  asm volatile("s_wait_asynccnt 0x0" ::: "memory");
}
__device__ __forceinline__ void wait_ds0() {
  asm volatile("s_wait_dscnt 0x0" ::: "memory");
}

// ---------------------------------------------------------------------------
// fp32 -> bf16 cast (vectorized x4)
// ---------------------------------------------------------------------------
__global__ __launch_bounds__(256) void cast_f32_bf16(const float* __restrict__ s,
                                                     __bf16* __restrict__ d, int n4) {
  int i = blockIdx.x * blockDim.x + threadIdx.x;
  if (i < n4) {
    const float4 v = ((const float4*)s)[i];
    v4bf o;
    o[0] = (__bf16)v.x; o[1] = (__bf16)v.y; o[2] = (__bf16)v.z; o[3] = (__bf16)v.w;
    ((v4bf*)d)[i] = o;
  }
}

// ---------------------------------------------------------------------------
// Tiled bf16 GEMM, fp32 accumulation, double-buffered async LDS staging:
//   Out[M,N] = alpha * A[M,K] @ B[N,K]^T + bias[N] + beta * Out[M,N]
// Block: 256 threads = 8 waves. Tile: BM=128 (16 rows/wave), BN=64 (4 n-tiles),
// BK=32 (one WMMA K per step). Fragment loads are contiguous b128 LDS reads
// with k = 16*hgrp + s for both A and B fragments. One barrier per K-step;
// next tile streams in via global_load_async_to_lds while WMMAs run.
// ---------------------------------------------------------------------------
#define BM 128
#define BN 64
#define BK 32

__global__ __launch_bounds__(256) void gemm_bf16_wmma(
    const __bf16* __restrict__ A,    // [M,K] row-major
    const __bf16* __restrict__ Bm,   // [N,K] row-major (i.e. B^T applied)
    const float*  __restrict__ bias, // [N] or nullptr
    void*         __restrict__ OutP, // [M,N] fp32 (or bf16 if out_bf16)
    int M, int N, int K, float alpha, float beta, int out_bf16)
{
  __shared__ __align__(64) __bf16 As[2][BM * BK];  // 2 x 8 KB
  __shared__ __align__(64) __bf16 Bs[2][BN * BK];  // 2 x 4 KB

  const int bm   = blockIdx.x * BM;
  const int bn   = blockIdx.y * BN;
  const int tid  = threadIdx.x;
  const int wave = tid >> 5;
  const int lane = tid & 31;
  const int hgrp = lane >> 4;   // lane half-group (0: lanes 0-15, 1: 16-31)
  const int l16  = lane & 15;

  // Per-thread staging coordinates
  const int ra  = tid >> 1;            // A row (0..127), 32B per thread
  const int kca = (tid & 1) * 16;
  const int rb  = tid >> 2;            // B row (0..63), 16B per thread
  const int kcb = (tid & 3) * 8;
  const int gnb = (bn + rb < N) ? (bn + rb) : 0;  // clamp (epilogue guards)

  auto stage = [&](int k0, int buf) {
    const __bf16* srcA = A + (size_t)(bm + ra) * K + k0 + kca;
    async_copy_b128(srcA,     &As[buf][ra * BK + kca]);
    async_copy_b128(srcA + 8, &As[buf][ra * BK + kca + 8]);
    const __bf16* srcB = Bm + (size_t)gnb * K + k0 + kcb;
    async_copy_b128(srcB, &Bs[buf][rb * BK + kcb]);
  };

  v8f acc[4] = {};
  const int nk = K / BK;

  stage(0, 0);
  wait_async0();
  __syncthreads();

  for (int i = 0; i < nk; ++i) {
    const int cur = i & 1;
    if (i + 1 < nk) stage((i + 1) * BK, 1 - cur);   // prefetch overlaps WMMA

    // A fragment (16x32): lane l16 = row m; K contiguous:
    // [m][hgrp*16 + 0..15] -> two 16B LDS loads.
    union { v16bf v; v8bf h8[2]; } af;
    const __bf16* arow = &As[cur][(wave * 16 + l16) * BK];
    af.h8[0] = *(const v8bf*)(arow + hgrp * 16);
    af.h8[1] = *(const v8bf*)(arow + hgrp * 16 + 8);

#pragma unroll
    for (int nt = 0; nt < 4; ++nt) {
      // B fragment (32x16): lane l16 = column n; K contiguous:
      // [n][hgrp*16 + 0..15] -> one 32B LDS load.
      const v16bf bfv = *(const v16bf*)&Bs[cur][(nt * 16 + l16) * BK + hgrp * 16];
      acc[nt] = __builtin_amdgcn_wmma_f32_16x16x32_bf16(
          false, af.v, false, bfv, (short)0, acc[nt], false, false);
    }

    wait_ds0();      // this wave's LDS reads retired (safe to overwrite later)
    wait_async0();   // this wave's prefetch landed
    __syncthreads(); // tile (i+1) complete for all waves
  }

  // Epilogue. C layout: VGPR r -> row (r + 8*hgrp), lane l16 -> column.
#pragma unroll
  for (int nt = 0; nt < 4; ++nt) {
    const int gn = bn + nt * 16 + l16;
    if (gn < N) {
      const float bv = bias ? bias[gn] : 0.0f;
#pragma unroll
      for (int r = 0; r < 8; ++r) {
        const int gm = bm + wave * 16 + r + hgrp * 8;
        float v = alpha * acc[nt][r] + bv;
        if (out_bf16) {
          ((__bf16*)OutP)[(size_t)gm * N + gn] = (__bf16)v;
        } else {
          float* p = (float*)OutP + (size_t)gm * N + gn;
          if (beta != 0.0f) v += beta * (*p);
          *p = v;
        }
      }
    }
  }
}

// ---------------------------------------------------------------------------
// Flash attention (causal), bf16 WMMA + online softmax, double-buffered K/V.
// qkv is pre-cast to bf16, so K tiles stage via async DMA (layout [kv][hd] is
// already the B-frag layout for q@k^T); V is transposed through VGPRs.
// Grid: (T/64, B*H). Block: 128 threads = 4 waves; wave owns 16 q rows.
// ---------------------------------------------------------------------------
#define QT 64
#define KT 32

__global__ __launch_bounds__(128) void flash_attn_wmma(
    const __bf16* __restrict__ qkv,  // [B*T, 3C] bf16
    __bf16*       __restrict__ y)    // [B*T, C]  bf16
{
  const int bh    = blockIdx.y;
  const int b     = bh >> 4;
  const int h     = bh & 15;
  const int qbase = blockIdx.x * QT;
  const int tid   = threadIdx.x;
  const int wave  = tid >> 5;
  const int lane  = tid & 31;
  const int hgrp  = lane >> 4;
  const int l16   = lane & 15;

  __shared__ __align__(64) __bf16 Ks[2][KT * HD];    // [kv][hd]  2 x 4 KB
  __shared__ __align__(64) __bf16 Vs[2][HD * KT];    // [hd][kv]  2 x 4 KB
  __shared__ __align__(64) __bf16 Ps[4][16 * KT];    // per-wave P, 4 KB

  const size_t rowbase = (size_t)b * TLEN;
  const int qcol = h * HD;
  const int kcol = CDIM + h * HD;
  const int vcol = 2 * CDIM + h * HD;

  // Staging coordinates: 32 kv rows, 16 elems (32B) per thread
  const int sr  = tid >> 2;
  const int seg = (tid & 3) * 16;

  auto stage = [&](int kv0, int buf) {
    const __bf16* kp = qkv + (rowbase + kv0 + sr) * C3 + kcol + seg;
    async_copy_b128(kp,     &Ks[buf][sr * HD + seg]);
    async_copy_b128(kp + 8, &Ks[buf][sr * HD + seg + 8]);
    const __bf16* vp = qkv + (rowbase + kv0 + sr) * C3 + vcol + seg;
    const v8bf v0 = *(const v8bf*)vp;
    const v8bf v1 = *(const v8bf*)(vp + 8);
#pragma unroll
    for (int e = 0; e < 8; ++e) Vs[buf][(seg + e) * KT + sr]     = v0[e];
#pragma unroll
    for (int e = 0; e < 8; ++e) Vs[buf][(seg + 8 + e) * KT + sr] = v1[e];
  };

  // Q fragments in registers: two 16x32 A-frags covering hd=64.
  union bfrag { v16bf v; v8bf h8[2]; };
  bfrag qf[2];
  {
    const int qr = qbase + wave * 16 + l16;
    const __bf16* qp = qkv + (rowbase + qr) * C3 + qcol;
#pragma unroll
    for (int c = 0; c < 2; ++c) {
      qf[c].h8[0] = *(const v8bf*)(qp + c * 32 + hgrp * 16);
      qf[c].h8[1] = *(const v8bf*)(qp + c * 32 + hgrp * 16 + 8);
    }
  }

  float rmax[8], rsum[8];
  v8f o[4] = {};
#pragma unroll
  for (int r = 0; r < 8; ++r) { rmax[r] = -1e30f; rsum[r] = 0.0f; }

  const int steps = (qbase + QT) / KT;   // causal bound for this q tile

  stage(0, 0);
  wait_ds0();
  wait_async0();
  __syncthreads();

  for (int i = 0; i < steps; ++i) {
    const int cur = i & 1;
    const int kv0 = i * KT;
    if (i + 1 < steps) stage((i + 1) * KT, 1 - cur);  // prefetch next K/V

    // S = q @ k^T : two 16x16 tiles (kv cols 0-15 / 16-31), 2 WMMAs each.
    v8f s[2] = {};
#pragma unroll
    for (int t = 0; t < 2; ++t) {
#pragma unroll
      for (int c = 0; c < 2; ++c) {
        const v16bf kb = *(const v16bf*)&Ks[cur][(t * 16 + l16) * HD + c * 32 + hgrp * 16];
        s[t] = __builtin_amdgcn_wmma_f32_16x16x32_bf16(
            false, qf[c].v, false, kb, (short)0, s[t], false, false);
      }
    }

    // Scale + causal mask.
    const int qrow_lo = qbase + wave * 16;
#pragma unroll
    for (int t = 0; t < 2; ++t) {
      const int col = kv0 + t * 16 + l16;
#pragma unroll
      for (int r = 0; r < 8; ++r) {
        const int row = qrow_lo + r + 8 * hgrp;
        s[t][r] = (col <= row) ? s[t][r] * 0.125f : -1e30f;
      }
    }

    // Online softmax: row max/sum via xor-reduction over the 16-lane
    // half-group holding each row (C layout: row r+8*hgrp, col l16).
    float scl[8];
#pragma unroll
    for (int r = 0; r < 8; ++r) {
      float m = fmaxf(s[0][r], s[1][r]);
#pragma unroll
      for (int d = 1; d < 16; d <<= 1) m = fmaxf(m, __shfl_xor(m, d, 32));
      const float nm = fmaxf(rmax[r], m);
      scl[r] = __expf(rmax[r] - nm);
      rmax[r] = nm;
    }
#pragma unroll
    for (int t = 0; t < 2; ++t)
#pragma unroll
      for (int r = 0; r < 8; ++r) s[t][r] = __expf(s[t][r] - rmax[r]);
#pragma unroll
    for (int r = 0; r < 8; ++r) {
      float su = s[0][r] + s[1][r];
#pragma unroll
      for (int d = 1; d < 16; d <<= 1) su += __shfl_xor(su, d, 32);
      rsum[r] = rsum[r] * scl[r] + su;
#pragma unroll
      for (int nt = 0; nt < 4; ++nt) o[nt][r] *= scl[r];
    }

    // C-layout -> A-layout for P via per-wave LDS bounce (same-wave LDS ops
    // are in-order; compiler inserts the RAW dscnt waits).
#pragma unroll
    for (int t = 0; t < 2; ++t)
#pragma unroll
      for (int r = 0; r < 8; ++r)
        Ps[wave][(r + 8 * hgrp) * KT + t * 16 + l16] = (__bf16)s[t][r];

    bfrag pf;
    const __bf16* prow = &Ps[wave][l16 * KT];
    pf.h8[0] = *(const v8bf*)(prow + hgrp * 16);
    pf.h8[1] = *(const v8bf*)(prow + hgrp * 16 + 8);

    // o += P[16x32] @ V[32x64] : 4 hd tiles, contiguous B-frags in Vs.
#pragma unroll
    for (int nt = 0; nt < 4; ++nt) {
      const v16bf vb = *(const v16bf*)&Vs[cur][(nt * 16 + l16) * KT + hgrp * 16];
      o[nt] = __builtin_amdgcn_wmma_f32_16x16x32_bf16(
          false, pf.v, false, vb, (short)0, o[nt], false, false);
    }

    wait_ds0();      // LDS reads + V-transpose stores retired
    wait_async0();   // prefetched K tile landed
    __syncthreads();
  }

  // Normalize and write y directly as bf16 (feeds the proj GEMM).
#pragma unroll
  for (int r = 0; r < 8; ++r) {
    const float inv = 1.0f / rsum[r];
    const int row = qbase + wave * 16 + r + 8 * hgrp;
    __bf16* yp = y + (rowbase + row) * CDIM + h * HD;
#pragma unroll
    for (int nt = 0; nt < 4; ++nt) yp[nt * 16 + l16] = (__bf16)(o[nt][r] * inv);
  }
}

// ---------------------------------------------------------------------------
// Host launcher
// ---------------------------------------------------------------------------
extern "C" void kernel_launch(void* const* d_in, const int* in_sizes, int n_in,
                              void* d_out, int out_size, void* d_ws, size_t ws_size,
                              hipStream_t stream) {
  (void)in_sizes; (void)n_in; (void)out_size; (void)ws_size;

  const float* x       = (const float*)d_in[0];
  const float* w_attn  = (const float*)d_in[1];
  const float* b_attn  = (const float*)d_in[2];
  const float* la_attn = (const float*)d_in[3];
  const float* lb_attn = (const float*)d_in[4];
  const float* w_proj  = (const float*)d_in[5];
  const float* b_proj  = (const float*)d_in[6];
  const float* la_proj = (const float*)d_in[7];
  const float* lb_proj = (const float*)d_in[8];
  float* out = (float*)d_out;

  // Workspace carve (256B aligned slots)
  char* ws = (char*)d_ws;
  auto take = [&](size_t bytes) { char* p = ws; ws += (bytes + 255) & ~(size_t)255; return p; };

  __bf16* x_bf     = (__bf16*)take((size_t)MROWS * CDIM * 2);
  __bf16* wattn_bf = (__bf16*)take((size_t)C3 * CDIM * 2);
  __bf16* laat_bf  = (__bf16*)take((size_t)RANK * CDIM * 2);
  __bf16* lbat_bf  = (__bf16*)take((size_t)C3 * RANK * 2);
  __bf16* wproj_bf = (__bf16*)take((size_t)CDIM * CDIM * 2);
  __bf16* lapr_bf  = (__bf16*)take((size_t)RANK * CDIM * 2);
  __bf16* lbpr_bf  = (__bf16*)take((size_t)CDIM * RANK * 2);
  float*  qkv      = (float*)take((size_t)MROWS * C3 * 4);
  __bf16* qkv_bf   = (__bf16*)take((size_t)MROWS * C3 * 2);
  __bf16* tmp_bf   = (__bf16*)take((size_t)MROWS * RANK * 2);
  __bf16* y_bf     = (__bf16*)take((size_t)MROWS * CDIM * 2);

  auto cast = [&](const float* s, __bf16* d, size_t n) {
    int n4 = (int)(n / 4);
    cast_f32_bf16<<<(n4 + 255) / 256, 256, 0, stream>>>(s, d, n4);
  };

  // 1) Casts of inputs / weights
  cast(x,       x_bf,     (size_t)MROWS * CDIM);
  cast(w_attn,  wattn_bf, (size_t)C3 * CDIM);
  cast(la_attn, laat_bf,  (size_t)RANK * CDIM);
  cast(lb_attn, lbat_bf,  (size_t)C3 * RANK);
  cast(w_proj,  wproj_bf, (size_t)CDIM * CDIM);
  cast(la_proj, lapr_bf,  (size_t)RANK * CDIM);
  cast(lb_proj, lbpr_bf,  (size_t)CDIM * RANK);

  dim3 blk(256);

  // 2) LoRA down (bf16 out): tmp = x @ la_attn^T   [4096 x 32]
  gemm_bf16_wmma<<<dim3(MROWS / BM, (RANK + BN - 1) / BN), blk, 0, stream>>>(
      x_bf, laat_bf, nullptr, tmp_bf, MROWS, RANK, CDIM, 1.0f, 0.0f, 1);

  // 3) qkv = x @ w_attn^T + b_attn ; then += scale * tmp @ lb_attn^T
  gemm_bf16_wmma<<<dim3(MROWS / BM, C3 / BN), blk, 0, stream>>>(
      x_bf, wattn_bf, b_attn, qkv, MROWS, C3, CDIM, 1.0f, 0.0f, 0);
  gemm_bf16_wmma<<<dim3(MROWS / BM, C3 / BN), blk, 0, stream>>>(
      tmp_bf, lbat_bf, nullptr, qkv, MROWS, C3, RANK, LORA_SCALE, 1.0f, 0);
  cast(qkv, qkv_bf, (size_t)MROWS * C3);

  // 4) Flash attention -> y_bf [4096 x 1024] (bf16, feeds proj GEMM)
  flash_attn_wmma<<<dim3(TLEN / QT, BSZ * NH), dim3(128), 0, stream>>>(qkv_bf, y_bf);

  // 5) Output projection with LoRA
  gemm_bf16_wmma<<<dim3(MROWS / BM, (RANK + BN - 1) / BN), blk, 0, stream>>>(
      y_bf, lapr_bf, nullptr, tmp_bf, MROWS, RANK, CDIM, 1.0f, 0.0f, 1);
  gemm_bf16_wmma<<<dim3(MROWS / BM, CDIM / BN), blk, 0, stream>>>(
      y_bf, wproj_bf, b_proj, out, MROWS, CDIM, CDIM, 1.0f, 0.0f, 0);
  gemm_bf16_wmma<<<dim3(MROWS / BM, CDIM / BN), blk, 0, stream>>>(
      tmp_bf, lbpr_bf, nullptr, out, MROWS, CDIM, RANK, LORA_SCALE, 1.0f, 0);
}
